// Q_Net_9594956939555
// MI455X (gfx1250) — matrix operand order repaired and checked by reading, not compiled, not run on hardware
//
#include <hip/hip_runtime.h>

typedef __attribute__((ext_vector_type(16))) _Float16 v16h;
typedef __attribute__((ext_vector_type(8)))  float    v8f;

#define BATCH   8192
#define CARD    8
#define NAGENT  64
#define HIDN    256
#define ENC_OUT 257
#define ACT     30
#define S_ROW   (CARD + NAGENT*CARD)   // 520
#define XPAD    288                    // Q-head K padded to 9*32
#define ENC_NT  17                     // 272 cols (257 padded)
#define Q1_KS   9
#define Q1_NT   16
#define Q2_NT   2

// workspace layout (bytes)
#define WS_W1   0u          // 16 blocks  * 1KB
#define WS_W2   16384u      // 136 blocks * 1KB
#define WS_QW1  155648u     // 144 blocks * 1KB
#define WS_QW2  303104u     // 16 blocks  * 1KB
#define WS_X    319488u     // 8192 * 288 f16

#define WMMA_F16(A, B, C) \
    __builtin_amdgcn_wmma_f32_16x16x32_f16(false, (A), false, (B), (short)0, (C), false, false)

// ---------------------------------------------------------------------------
// Prep: f32 -> f16 weight conversion + pre-swizzle into WMMA B-fragment order.
// Block = (n_tile * k_steps + k_step); inside a block: [lane][half 0..15].
// B order used by this code (contiguous k): lane n (0..15) holds col n, K = h; lanes 16..31
// hold col n-16, K = 16 + h  =>  k = k_step*32 + (lane>>4)*16 + h.
// ---------------------------------------------------------------------------
__global__ void swizzle_w(const float* __restrict__ W, _Float16* __restrict__ dst,
                          int k_steps, int K_valid, int N_valid, int ldw, int total)
{
    int tid = blockIdx.x * 256 + threadIdx.x;
    if (tid >= total) return;
    int blk  = tid >> 9;
    int rem  = tid & 511;
    int lane = rem >> 4;
    int h    = rem & 15;
    int n_tile = blk / k_steps;
    int k_step = blk - n_tile * k_steps;
    int n = n_tile * 16 + (lane & 15);
    int k = k_step * 32 + ((lane >> 4) << 4) + h;
    float v = (k < K_valid && n < N_valid) ? W[(size_t)k * ldw + n] : 0.0f;
    dst[tid] = (_Float16)v;
}

// ---------------------------------------------------------------------------
// A-fragment gather (16-bit A, 16x32): lane l holds row m = l%16;
// VGPR v, half p: k = (l>>4)*16 + v*2 + p  (pairs contiguous).
// ---------------------------------------------------------------------------
union AFrag { v16h v; unsigned u[8]; };

__device__ __forceinline__ v16h gather_a(const _Float16* rowbase, int lane, int kbase)
{
    AFrag af;
#pragma unroll
    for (int vi = 0; vi < 8; ++vi) {
        int k0 = kbase + ((lane >> 4) << 4) + (vi << 1);
        af.u[vi] = *(const unsigned*)(rowbase + k0);
    }
    return af.v;
}

__device__ __forceinline__ float frelu(float x) { return x > 0.0f ? x : 0.0f; }

// ---------------------------------------------------------------------------
// Encoder: one WAVE per batch element (4 elements per 128-thread WG).
// The wave keeps all A fragments (4 M-tiles x 8 K-steps) in VGPRs and loads
// each B fragment ONCE per n-tile, reusing it across its 4 M-tiles -> 4x less
// L2 traffic than one-element-per-WG.  All LDS state is wave-private, so no
// block barriers are needed.
// ---------------------------------------------------------------------------
__global__ void __launch_bounds__(128)
encoder_kernel(const float* __restrict__ s,
               const _Float16* __restrict__ W1s,
               const float* __restrict__ b1,
               const _Float16* __restrict__ W2s,
               const float* __restrict__ b2,
               _Float16* __restrict__ xout)
{
    __shared__ __align__(16) float surr[4][NAGENT * CARD];    // 8 KB
    __shared__ float validf[4][NAGENT];                       // 1 KB
    __shared__ __align__(4) _Float16 hbuf[4][NAGENT * HIDN];  // 128 KB
    __shared__ float sums[4][ENC_NT * 16];                    // 4.25 KB

    const int lane = threadIdx.x & 31;
    const int w    = threadIdx.x >> 5;
    const int b    = blockIdx.x * 4 + w;
    const float* srow = s + (size_t)b * S_ROW;

    // ---- stage this element's 64x8 surrounding block (512 floats)
    {
        const float4* src = (const float4*)(srow + CARD);
        float4* dst = (float4*)surr[w];
#pragma unroll
        for (int i = 0; i < 4; ++i)
            dst[lane + 32 * i] = src[lane + 32 * i];
    }
#pragma unroll
    for (int i = lane; i < ENC_NT * 16; i += 32) sums[w][i] = 0.0f;
#pragma unroll
    for (int ag = lane; ag < NAGENT; ag += 32) {
        bool ok = true;
#pragma unroll
        for (int c = 0; c < CARD; ++c) ok &= (surr[w][ag * CARD + c] != -1.0f);
        validf[w][ag] = ok ? 1.0f : 0.0f;
    }

    // ---- layer 1: A = surr rows (K=8 zero-padded to 32), 4 M-tiles per wave
    v16h a1[4];
#pragma unroll
    for (int mt = 0; mt < 4; ++mt) {
        int m = mt * 16 + (lane & 15);
#pragma unroll
        for (int i = 0; i < 16; ++i) {
            float sv = surr[w][m * CARD + (i & 7)];
            a1[mt][i] = (_Float16)((lane < 16 && i < 8) ? sv : 0.0f);
        }
    }
    const v16h* W1f = (const v16h*)W1s;
#pragma unroll 1
    for (int nt = 0; nt < 16; ++nt) {
        v16h bf = W1f[nt * 32 + lane];
        int n = nt * 16 + (lane & 15);
        float bias = b1[n];
#pragma unroll
        for (int mt = 0; mt < 4; ++mt) {
            v8f c = {};
            c = WMMA_F16(a1[mt], bf, c);
            int m0 = mt * 16 + ((lane >> 4) << 3);
#pragma unroll
            for (int r = 0; r < 8; ++r)
                hbuf[w][(m0 + r) * HIDN + n] = (_Float16)frelu(c[r] + bias);
        }
    }

    // ---- layer 2: pre-gather ALL A fragments (4 mt x 8 ks) into VGPRs
    v16h a2[4][8];
#pragma unroll
    for (int mt = 0; mt < 4; ++mt) {
        const _Float16* rowb = &hbuf[w][(mt * 16 + (lane & 15)) * HIDN];
#pragma unroll
        for (int ks = 0; ks < 8; ++ks) a2[mt][ks] = gather_a(rowb, lane, ks * 32);
    }
    float vmask[4][8];
#pragma unroll
    for (int mt = 0; mt < 4; ++mt) {
        int m0 = mt * 16 + ((lane >> 4) << 3);
#pragma unroll
        for (int r = 0; r < 8; ++r) vmask[mt][r] = validf[w][m0 + r];
    }

    const v16h* W2f = (const v16h*)W2s;
#pragma unroll 1
    for (int nt = 0; nt < ENC_NT; ++nt) {
        v16h bf[8];
#pragma unroll
        for (int ks = 0; ks < 8; ++ks) bf[ks] = W2f[(nt * 8 + ks) * 32 + lane];
        if (nt + 1 < ENC_NT)   // overlap next tile's L2 fetch with WMMA chain
            __builtin_prefetch(&W2f[(nt + 1) * 8 * 32 + lane], 0, 1);

        int n = nt * 16 + (lane & 15);
        float bias = (n < ENC_OUT) ? b2[n] : 0.0f;
        float part = 0.0f;
#pragma unroll
        for (int mt = 0; mt < 4; ++mt) {
            v8f c = {};
#pragma unroll
            for (int ks = 0; ks < 8; ++ks) c = WMMA_F16(a2[mt][ks], bf[ks], c);
#pragma unroll
            for (int r = 0; r < 8; ++r)
                part += frelu(c[r] + bias) * vmask[mt][r];
        }
        atomicAdd(&sums[w][n], part);   // lane n vs n+16 pair only
    }

    // ---- emit x row = [self(8) | sum_outputs(257) | zero pad] as f16
    _Float16* xr = xout + (size_t)b * XPAD;
#pragma unroll
    for (int idx = lane; idx < XPAD; idx += 32) {
        float v;
        if (idx < CARD)                 v = srow[idx];
        else if (idx < CARD + ENC_OUT)  v = sums[w][idx - CARD];
        else                            v = 0.0f;
        xr[idx] = (_Float16)v;
    }
}

// ---------------------------------------------------------------------------
// Q-head: each wave owns 16 batch rows.  q = relu(x@Qw1+Qb1)@Qw2 + Qb2.
// ---------------------------------------------------------------------------
__global__ void __launch_bounds__(128)
qhead_kernel(const _Float16* __restrict__ xin,
             const _Float16* __restrict__ Qw1s,
             const float* __restrict__ qb1,
             const _Float16* __restrict__ Qw2s,
             const float* __restrict__ qb2,
             float* __restrict__ out)
{
    __shared__ __align__(4) _Float16 hq[4][16 * HIDN];    // 4 x 8KB
    const int lane = threadIdx.x & 31;
    const int w    = threadIdx.x >> 5;
    const int tile = blockIdx.x * 4 + w;

    v16h a[Q1_KS];
    {
        const _Float16* rowb = xin + (size_t)(tile * 16 + (lane & 15)) * XPAD;
#pragma unroll
        for (int ks = 0; ks < Q1_KS; ++ks) a[ks] = gather_a(rowb, lane, ks * 32);
    }
    const v16h* W1f = (const v16h*)Qw1s;
#pragma unroll 1
    for (int nt = 0; nt < Q1_NT; ++nt) {
        if (nt + 1 < Q1_NT)
            __builtin_prefetch(&W1f[(nt + 1) * Q1_KS * 32 + lane], 0, 1);
        v8f c = {};
#pragma unroll
        for (int ks = 0; ks < Q1_KS; ++ks) {
            v16h bfrag = W1f[(nt * Q1_KS + ks) * 32 + lane];
            c = WMMA_F16(a[ks], bfrag, c);
        }
        int n = nt * 16 + (lane & 15);
        float bias = qb1[n];
        int m0 = (lane >> 4) << 3;
#pragma unroll
        for (int r = 0; r < 8; ++r)
            hq[w][(m0 + r) * HIDN + n] = (_Float16)frelu(c[r] + bias);
    }

    v16h a2[8];
    {
        const _Float16* rowb = &hq[w][(lane & 15) * HIDN];
#pragma unroll
        for (int ks = 0; ks < 8; ++ks) a2[ks] = gather_a(rowb, lane, ks * 32);
    }
    const v16h* W2f = (const v16h*)Qw2s;
#pragma unroll 1
    for (int nt = 0; nt < Q2_NT; ++nt) {
        v8f c = {};
#pragma unroll
        for (int ks = 0; ks < 8; ++ks) {
            v16h bfrag = W2f[(nt * 8 + ks) * 32 + lane];
            c = WMMA_F16(a2[ks], bfrag, c);
        }
        int n = nt * 16 + (lane & 15);
        int m0 = tile * 16 + ((lane >> 4) << 3);
        if (n < ACT) {
            float bias = qb2[n];
#pragma unroll
            for (int r = 0; r < 8; ++r)
                out[(size_t)(m0 + r) * ACT + n] = c[r] + bias;
        }
    }
}

// ---------------------------------------------------------------------------
extern "C" void kernel_launch(void* const* d_in, const int* in_sizes, int n_in,
                              void* d_out, int out_size, void* d_ws, size_t ws_size,
                              hipStream_t stream)
{
    (void)in_sizes; (void)n_in; (void)out_size; (void)ws_size;
    const float* s   = (const float*)d_in[0];
    const float* W1  = (const float*)d_in[1];
    const float* b1  = (const float*)d_in[2];
    const float* W2  = (const float*)d_in[3];
    const float* b2  = (const float*)d_in[4];
    const float* Qw1 = (const float*)d_in[5];
    const float* Qb1 = (const float*)d_in[6];
    const float* Qw2 = (const float*)d_in[7];
    const float* Qb2 = (const float*)d_in[8];

    char* ws = (char*)d_ws;
    _Float16* W1s  = (_Float16*)(ws + WS_W1);
    _Float16* W2s  = (_Float16*)(ws + WS_W2);
    _Float16* Qw1s = (_Float16*)(ws + WS_QW1);
    _Float16* Qw2s = (_Float16*)(ws + WS_QW2);
    _Float16* X    = (_Float16*)(ws + WS_X);

    // weight swizzle: (W, dst, k_steps, K_valid, N_valid, ldw, total halves)
    swizzle_w<<<(16 * 512) / 256,     256, 0, stream>>>(W1,  W1s,  1, 8,   256, 256, 16 * 512);
    swizzle_w<<<(17 * 8 * 512) / 256, 256, 0, stream>>>(W2,  W2s,  8, 256, 257, 257, 17 * 8 * 512);
    swizzle_w<<<(16 * 9 * 512) / 256, 256, 0, stream>>>(Qw1, Qw1s, 9, 265, 256, 256, 16 * 9 * 512);
    swizzle_w<<<(2 * 8 * 512) / 256,  256, 0, stream>>>(Qw2, Qw2s, 8, 256, 30,  30,  2 * 8 * 512);

    encoder_kernel<<<BATCH / 4, 128, 0, stream>>>(s, W1s, b1, W2s, b2, X);
    qhead_kernel<<<BATCH / 64, 128, 0, stream>>>(X, Qw1s, Qb1, Qw2s, Qb2, (float*)d_out);
}
